// SingleHeadAttention_57612691308546
// MI455X (gfx1250) — matrix operand order repaired and checked by reading, not compiled, not run on hardware
//
#include <hip/hip_runtime.h>

// B=4, S=2048, D=1024 single-head causal attention, fp32 in/out.
// All four GEMMs run on v_wmma_f32_16x16x32_f16 (f32 accumulate).
// f16 chosen over bf16: every tensor here (x, W, q/k/v, scores<~200, P in [0,1])
// fits f16 range, and f16's 10 mantissa bits beat bf16's 7 at identical WMMA
// throughput. f16 scores also halve the LDS score buffer, making room for a
// separate P buffer -> fully parallel (8 threads/row) softmax.

typedef _Float16 f16;
typedef __attribute__((ext_vector_type(16))) _Float16 v16h;
typedef __attribute__((ext_vector_type(8)))  float    v8f;

union FragAB { v16h v; uint4 u[2]; };
union Pack8  { f16 h[8]; uint4 u; };
union Pack4  { f16 h[4]; uint2 u; };
union Pack2  { f16 h[2]; unsigned u; };

#define NEG_INF (-__builtin_inff())
#define WMMA_F16(a,b,c) __builtin_amdgcn_wmma_f32_16x16x32_f16(false,(a),false,(b),(short)0,(c),false,false)

__device__ __forceinline__ v8f zero_v8f() {
    v8f z = {0.f,0.f,0.f,0.f,0.f,0.f,0.f,0.f};
    return z;
}

// ---------------------------------------------------------------------------
// Kernel 0: fp32 -> f16 conversion (vectorized x4)
// ---------------------------------------------------------------------------
__global__ void cvt_f32_f16(const float* __restrict__ in, f16* __restrict__ out, int n4) {
    int i = blockIdx.x * blockDim.x + threadIdx.x;
    if (i >= n4) return;
    float4 v = ((const float4*)in)[i];
    Pack4 p;
    p.h[0] = (f16)v.x; p.h[1] = (f16)v.y; p.h[2] = (f16)v.z; p.h[3] = (f16)v.w;
    ((uint2*)out)[i] = p.u;
}

// ---------------------------------------------------------------------------
// Kernel 1: QKV projection.  C[M=8192, N=1024] = Xh @ W^T (W row-major [out,in],
// so B-frag lane n reads W[n, k..k+15] contiguously).  mode 0/1 -> Q/K row-major
// f16; mode 2 -> V stored transposed per batch as VT[b][d][s] (contiguous 16B
// store per lane thanks to the C-matrix VGPR layout).
// ---------------------------------------------------------------------------
__global__ __launch_bounds__(256)
void qkv_gemm(const f16* __restrict__ X,  const f16* __restrict__ Wq,
              const f16* __restrict__ Wk, const f16* __restrict__ Wv,
              f16* __restrict__ Q, f16* __restrict__ Kout, f16* __restrict__ VT) {
    const int mode = blockIdx.z;
    const f16* W = (mode == 0) ? Wq : (mode == 1) ? Wk : Wv;
    const int lane = threadIdx.x & 31;
    const int wave = threadIdx.x >> 5;
    const int wm = wave & 3;            // 4 waves down
    const int wn = wave >> 2;           // 2 waves across
    const int mbase = blockIdx.y * 128 + wm * 32;
    const int nbase = blockIdx.x * 128 + wn * 64;
    const int r  = lane & 15;
    const int hh = lane >> 4;

    v8f acc[2][4];
#pragma unroll
    for (int mt = 0; mt < 2; ++mt)
#pragma unroll
        for (int nt = 0; nt < 4; ++nt) acc[mt][nt] = zero_v8f();

    for (int kb = 0; kb < 1024; kb += 32) {
        FragAB a[2], b[4];
#pragma unroll
        for (int mt = 0; mt < 2; ++mt) {
            const f16* p = X + (size_t)(mbase + mt * 16 + r) * 1024 + kb + hh * 16;
            a[mt].u[0] = *(const uint4*)p;          // K kh*16 + 0..7
            a[mt].u[1] = *(const uint4*)(p + 8);    // K kh*16 + 8..15
        }
#pragma unroll
        for (int nt = 0; nt < 4; ++nt) {
            const f16* p = W + (size_t)(nbase + nt * 16 + r) * 1024 + kb + hh * 16;
            b[nt].u[0] = *(const uint4*)p;          // K kh*16 + 0..7
            b[nt].u[1] = *(const uint4*)(p + 8);    // K kh*16 + 8..15
        }
#pragma unroll
        for (int mt = 0; mt < 2; ++mt)
#pragma unroll
            for (int nt = 0; nt < 4; ++nt)
                acc[mt][nt] = WMMA_F16(a[mt].v, b[nt].v, acc[mt][nt]);
    }

    if (mode < 2) {
        f16* Out = mode ? Kout : Q;
#pragma unroll
        for (int mt = 0; mt < 2; ++mt)
#pragma unroll
            for (int nt = 0; nt < 4; ++nt)
#pragma unroll
                for (int e = 0; e < 8; ++e) {
                    int row = mbase + mt * 16 + hh * 8 + e;
                    Out[(size_t)row * 1024 + nbase + nt * 16 + r] = (f16)acc[mt][nt][e];
                }
    } else {
        // VT[b][d][s]: lane holds fixed d, 8 consecutive s -> one b128 store.
#pragma unroll
        for (int mt = 0; mt < 2; ++mt)
#pragma unroll
            for (int nt = 0; nt < 4; ++nt) {
                int mg = mbase + mt * 16 + hh * 8;   // global row = b*2048 + s
                int bb = mg >> 11;
                int s  = mg & 2047;
                int d  = nbase + nt * 16 + r;
                Pack8 pk;
#pragma unroll
                for (int e = 0; e < 8; ++e) pk.h[e] = (f16)acc[mt][nt][e];
                *(uint4*)(VT + ((size_t)bb * 1024 + d) * 2048 + s) = pk.u;
            }
    }
}

// ---------------------------------------------------------------------------
// Kernel 2: causal attention, one 32-row query block per workgroup.
//   Phase A: scores[32, kend] f16 into LDS (8 waves split the 16-wide j tiles),
//            causal mask applied at the write.
//   Phase B: parallel softmax, 8 threads per row (segmented max / exp+sum with
//            LDS partial combine); P written f16 to a separate buffer.
//   Phase C: out = (P @ V) / rowsum; A-frags from LDS P, B-frags from VT.
// LDS: scores 32x4112B + P 32x4112B (4-bank row skew, conflict-free b128 frag
// reads) + partial-stat scratch  =  ~259 KB of the 320 KB WGP budget.
// ---------------------------------------------------------------------------
#define ROWB 4112                       // 2048 f16 + 16B pad per row
#define POFF (32 * ROWB)                // P buffer offset
#define SOFF (2 * 32 * ROWB)            // stats offset

__global__ __launch_bounds__(256)
void attention(const f16* __restrict__ Q, const f16* __restrict__ Km,
               const f16* __restrict__ VT, float* __restrict__ out) {
    extern __shared__ char smem[];
    float* pmax   = (float*)(smem + SOFF);            // [32][8]
    float* psum   = (float*)(smem + SOFF + 1024);     // [32][8]
    float* rowsum = (float*)(smem + SOFF + 2048);     // [32]

    const int b    = blockIdx.y;
    const int qb   = blockIdx.x;
    const int m0   = qb * 32;
    const int kend = m0 + 32;           // causal: keys 0..m0+31, multiple of 32
    const int tid  = threadIdx.x;
    const int lane = tid & 31;
    const int wave = tid >> 5;
    const int r    = lane & 15;
    const int hh   = lane >> 4;
    const float scale = 0.03125f;       // 1/sqrt(1024)

    const f16* Qb = Q  + (size_t)b * 2048 * 1024;
    const f16* Kb = Km + (size_t)b * 2048 * 1024;
    const f16* Vb = VT + (size_t)b * 1024 * 2048;

    // ---------------- Phase A: scores = Q K^T ----------------
    const int ntile = kend >> 4;        // 16-wide j tiles
    for (int t = wave; t < ntile; t += 8) {
        const int jb = t * 16;
        v8f acc[2];
        acc[0] = zero_v8f(); acc[1] = zero_v8f();
        for (int kb = 0; kb < 1024; kb += 32) {
            FragAB a0, a1, bb;
            const f16* pa0 = Qb + (size_t)(m0 + r) * 1024 + kb + hh * 16;
            a0.u[0] = *(const uint4*)pa0;
            a0.u[1] = *(const uint4*)(pa0 + 8);
            const f16* pa1 = pa0 + (size_t)16 * 1024;
            a1.u[0] = *(const uint4*)pa1;
            a1.u[1] = *(const uint4*)(pa1 + 8);
            const f16* pb = Kb + (size_t)(jb + r) * 1024 + kb + hh * 16;
            bb.u[0] = *(const uint4*)pb;
            bb.u[1] = *(const uint4*)(pb + 8);
            acc[0] = WMMA_F16(a0.v, bb.v, acc[0]);
            acc[1] = WMMA_F16(a1.v, bb.v, acc[1]);
        }
#pragma unroll
        for (int mt = 0; mt < 2; ++mt)
#pragma unroll
            for (int e = 0; e < 8; ++e) {
                int ml = mt * 16 + hh * 8 + e;      // local query row 0..31
                int j  = jb + r;
                float v = (j <= m0 + ml) ? acc[mt][e] : NEG_INF;
                *(f16*)(smem + (size_t)ml * ROWB + (size_t)j * 2) = (f16)v;
            }
    }
    __syncthreads();

    // ---------------- Phase B: parallel softmax (8 threads / row) ----------
    {
        const int row = tid >> 3;       // 0..31
        const int sub = tid & 7;        // 0..7
        const int seg = kend >> 3;      // multiple of 4
        const int s0  = sub * seg;
        const f16* sr = (const f16*)(smem + (size_t)row * ROWB);

        float mx = NEG_INF;
        for (int k = s0; k < s0 + seg; k += 2) {
            Pack2 pk; pk.u = *(const unsigned*)(sr + k);
            mx = fmaxf(mx, fmaxf((float)pk.h[0], (float)pk.h[1]));
        }
        pmax[row * 8 + sub] = mx;
        __syncthreads();

        float gmx = pmax[row * 8];
#pragma unroll
        for (int j = 1; j < 8; ++j) gmx = fmaxf(gmx, pmax[row * 8 + j]);

        f16* pr = (f16*)(smem + POFF + (size_t)row * ROWB);
        float sum = 0.f;
        for (int k = s0; k < s0 + seg; k += 2) {
            Pack2 pk; pk.u = *(const unsigned*)(sr + k);
            float e0 = __expf(((float)pk.h[0] - gmx) * scale);
            float e1 = __expf(((float)pk.h[1] - gmx) * scale);
            sum += e0 + e1;
            Pack2 po; po.h[0] = (f16)e0; po.h[1] = (f16)e1;
            *(unsigned*)(pr + k) = po.u;
        }
        psum[row * 8 + sub] = sum;
        __syncthreads();

        if (sub == 0) {
            float s = 0.f;
#pragma unroll
            for (int j = 0; j < 8; ++j) s += psum[row * 8 + j];
            rowsum[row] = s;
        }
    }
    __syncthreads();

    // ---------------- Phase C: out = (P @ V) / rowsum ----------------
    for (int pass = 0; pass < 2; ++pass) {
        const int nc0 = pass * 512 + wave * 64;
        v8f acc[2][4];
#pragma unroll
        for (int mt = 0; mt < 2; ++mt)
#pragma unroll
            for (int nt = 0; nt < 4; ++nt) acc[mt][nt] = zero_v8f();

        for (int kb = 0; kb < kend; kb += 32) {
            FragAB a[2], bb[4];
#pragma unroll
            for (int mt = 0; mt < 2; ++mt) {
                const char* pr = smem + POFF + (size_t)(mt * 16 + r) * ROWB
                               + (size_t)(kb + hh * 16) * 2;
                a[mt].u[0] = *(const uint4*)pr;
                a[mt].u[1] = *(const uint4*)(pr + 16);
            }
#pragma unroll
            for (int nt = 0; nt < 4; ++nt) {
                const f16* p = Vb + (size_t)(nc0 + nt * 16 + r) * 2048 + kb + hh * 16;
                bb[nt].u[0] = *(const uint4*)p;
                bb[nt].u[1] = *(const uint4*)(p + 8);
            }
#pragma unroll
            for (int mt = 0; mt < 2; ++mt)
#pragma unroll
                for (int nt = 0; nt < 4; ++nt)
                    acc[mt][nt] = WMMA_F16(a[mt].v, bb[nt].v, acc[mt][nt]);
        }
#pragma unroll
        for (int mt = 0; mt < 2; ++mt)
#pragma unroll
            for (int e = 0; e < 8; ++e) {
                int ml = mt * 16 + hh * 8 + e;
                float inv = 1.f / rowsum[ml];
#pragma unroll
                for (int nt = 0; nt < 4; ++nt)
                    out[(size_t)(b * 2048 + m0 + ml) * 1024 + nc0 + nt * 16 + r]
                        = acc[mt][nt][e] * inv;
            }
    }
}

// ---------------------------------------------------------------------------
extern "C" void kernel_launch(void* const* d_in, const int* in_sizes, int n_in,
                              void* d_out, int out_size, void* d_ws, size_t ws_size,
                              hipStream_t stream) {
    const float* x  = (const float*)d_in[0];   // [4,2048,1024]
    const float* Wq = (const float*)d_in[1];   // [1024,1024]
    const float* Wk = (const float*)d_in[2];
    const float* Wv = (const float*)d_in[3];

    char* ws = (char*)d_ws;
    f16* xh  = (f16*)(ws);                        // 16 MB
    f16* wqh = (f16*)(ws + (size_t)16 * 1048576); //  2 MB
    f16* wkh = (f16*)(ws + (size_t)18 * 1048576); //  2 MB
    f16* wvh = (f16*)(ws + (size_t)20 * 1048576); //  2 MB
    f16* Qh  = (f16*)(ws + (size_t)22 * 1048576); // 16 MB
    f16* Kh  = (f16*)(ws + (size_t)38 * 1048576); // 16 MB
    f16* VTh = (f16*)(ws + (size_t)54 * 1048576); // 16 MB (total 70 MB)

    // fp32 -> f16
    cvt_f32_f16<<<8192, 256, 0, stream>>>(x,  xh,  (4 * 2048 * 1024) / 4);
    cvt_f32_f16<<<1024, 256, 0, stream>>>(Wq, wqh, (1024 * 1024) / 4);
    cvt_f32_f16<<<1024, 256, 0, stream>>>(Wk, wkh, (1024 * 1024) / 4);
    cvt_f32_f16<<<1024, 256, 0, stream>>>(Wv, wvh, (1024 * 1024) / 4);

    // Q/K row-major, V transposed; grid.z selects which projection.
    qkv_gemm<<<dim3(8, 64, 3), 256, 0, stream>>>(xh, wqh, wkh, wvh, Qh, Kh, VTh);

    // 32 query rows per workgroup; ~259 KB dynamic LDS.
    attention<<<dim3(64, 4), 256, SOFF + 2048 + 128, stream>>>(Qh, Kh, VTh, (float*)d_out);
}
